// GraphTransformer_63531156242782
// MI455X (gfx1250) — compile-verified, matrix operand order checked by reading, not run on hardware
//
#include <hip/hip_runtime.h>
#include <hip/hip_bf16.h>
#include <math.h>

// ---------------- problem constants ----------------
#define N_NODES 2048
#define DMODEL  256
#define FDIM    1024
#define NLAYER  4
#define NCLS    10
#define NHEAD   8
#define DHEAD   32

typedef __attribute__((ext_vector_type(16))) __bf16 bf16x16;
typedef __attribute__((ext_vector_type(8)))  __bf16 bf16x8;
typedef __attribute__((ext_vector_type(8)))  float  f32x8;
typedef int v4i_vs __attribute__((vector_size(16)));   // matches builtin param type

#define AS1 __attribute__((address_space(1)))
#define AS3 __attribute__((address_space(3)))

#if defined(__gfx1250__) && __has_builtin(__builtin_amdgcn_global_load_async_to_lds_b128)
#define USE_ASYNC_LDS 1
#else
#define USE_ASYNC_LDS 0
#endif

__device__ __forceinline__ void wait_async_lds() {
#if USE_ASYNC_LDS
#if __has_builtin(__builtin_amdgcn_s_wait_asynccnt)
  __builtin_amdgcn_s_wait_asynccnt(0);
#else
  asm volatile("s_wait_asynccnt 0x0" ::: "memory");
#endif
#endif
}

// stage one 16-byte chunk global -> LDS
__device__ __forceinline__ void stage_chunk(const __bf16* gsrc, __bf16* ldst) {
#if USE_ASYNC_LDS
  __builtin_amdgcn_global_load_async_to_lds_b128(
      (AS1 v4i_vs*)const_cast<__bf16*>(gsrc), (AS3 v4i_vs*)ldst, 0, 0);
#else
  *(uint4*)ldst = *(const uint4*)gsrc;
#endif
}

__device__ __forceinline__ __bf16 f2bf(float f) {
  unsigned u = __builtin_bit_cast(unsigned, f);
  u += 0x7FFFu + ((u >> 16) & 1u);           // round-to-nearest-even
  unsigned short h = (unsigned short)(u >> 16);
  return __builtin_bit_cast(__bf16, h);
}

__device__ __forceinline__ float gelu_exact(float x) {
  return 0.5f * x * (1.0f + erff(x * 0.70710678118654752f));
}

// ---------------- WMMA NT GEMM: C(M x Nc) = A(M x K) * B(Nc x K)^T ----------------
// 128x64 block tile, 8 waves, each wave computes a 32x32 C slab (4 WMMAs / K-step).
// Double-buffered LDS filled by GLOBAL_LOAD_ASYNC_TO_LDS_B128 (ASYNCcnt).
#define TM 128
#define TN 64
#define TK 32
#define LDP 40                 // padded LDS row stride (bf16): 80 B, 16 B aligned rows
#define ASZ (TM * LDP)         // 5120 bf16
#define BSZ (TN * LDP)         // 2560 bf16
#define BUFSZ (ASZ + BSZ)      // per-buffer elems (15360 B)

enum { EP_F32 = 0, EP_BF16 = 1, EP_ADD_F32 = 2, EP_GELU_BF16 = 3, EP_QKV = 4 };

__global__ __launch_bounds__(256) void gemm_nt_bf16_wmma(
    const __bf16* __restrict__ A, const __bf16* __restrict__ B,
    int M, int Nc, int K, int mode,
    float* __restrict__ outF, const float* __restrict__ bias,
    int ldo, int col0,
    __bf16* __restrict__ ob0, __bf16* __restrict__ ob1, __bf16* __restrict__ ob2)
{
  __shared__ __bf16 smem[2 * BUFSZ];

  const int t   = threadIdx.x;
  const int bm0 = blockIdx.y * TM;
  const int bn0 = blockIdx.x * TN;

  // staging: per thread, two A chunks (rows r, r+64) and one B chunk, 16 B each
  const int sr = t >> 2;          // 0..63
  const int sc = (t & 3) * 8;     // 0,8,16,24
  int gra0 = bm0 + sr;        if (gra0 >= M)  gra0 = M - 1;   // clamp: garbage lands
  int gra1 = bm0 + 64 + sr;   if (gra1 >= M)  gra1 = M - 1;   // only in discarded C
  int grb  = bn0 + sr;        if (grb  >= Nc) grb  = Nc - 1;

  // wave layout: 8 waves -> 4x2 grid of 32x32 C slabs
  const int w  = t >> 5;
  const int l  = t & 31;
  const int mt = (w & 3) * 32;
  const int nt = (w >> 2) * 32;
  const int lm = l & 15;
  const int kh = l >> 4;

  f32x8 acc00 = {0.f,0.f,0.f,0.f,0.f,0.f,0.f,0.f};
  f32x8 acc01 = acc00, acc10 = acc00, acc11 = acc00;

  const int nk = K / TK;

  auto issue = [&](int kt, int buf) {
    __bf16* tA = smem + buf * BUFSZ;
    __bf16* tB = tA + ASZ;
    const size_t ko = (size_t)kt * TK + sc;
    stage_chunk(A + (size_t)gra0 * K + ko, &tA[sr * LDP + sc]);
    stage_chunk(A + (size_t)gra1 * K + ko, &tA[(64 + sr) * LDP + sc]);
    stage_chunk(B + (size_t)grb  * K + ko, &tB[sr * LDP + sc]);
  };

  issue(0, 0);

  for (int kt = 0; kt < nk; ++kt) {
    const int cur = kt & 1;
    wait_async_lds();
    __syncthreads();                       // tile `cur` ready for everyone
    if (kt + 1 < nk) issue(kt + 1, cur ^ 1);  // overlap DMA with compute

    __bf16* tA = smem + cur * BUFSZ;
    __bf16* tB = tA + ASZ;

    union { bf16x16 v; bf16x8 h[2]; } af0, af1, bv0, bv1;
    const __bf16* pa0 = &tA[(mt + lm) * LDP];
    af0.h[0] = *(const bf16x8*)(pa0 + kh * 8);
    af0.h[1] = *(const bf16x8*)(pa0 + 16 + kh * 8);
    const __bf16* pa1 = &tA[(mt + 16 + lm) * LDP];
    af1.h[0] = *(const bf16x8*)(pa1 + kh * 8);
    af1.h[1] = *(const bf16x8*)(pa1 + 16 + kh * 8);
    const __bf16* pb0 = &tB[(nt + lm) * LDP];
    bv0.h[0] = *(const bf16x8*)(pb0 + kh * 8);
    bv0.h[1] = *(const bf16x8*)(pb0 + 16 + kh * 8);
    const __bf16* pb1 = &tB[(nt + 16 + lm) * LDP];
    bv1.h[0] = *(const bf16x8*)(pb1 + kh * 8);
    bv1.h[1] = *(const bf16x8*)(pb1 + 16 + kh * 8);

    acc00 = __builtin_amdgcn_wmma_f32_16x16x32_bf16(false, af0.v, false, bv0.v, (short)0, acc00, false, false);
    acc01 = __builtin_amdgcn_wmma_f32_16x16x32_bf16(false, af0.v, false, bv1.v, (short)0, acc01, false, false);
    acc10 = __builtin_amdgcn_wmma_f32_16x16x32_bf16(false, af1.v, false, bv0.v, (short)0, acc10, false, false);
    acc11 = __builtin_amdgcn_wmma_f32_16x16x32_bf16(false, af1.v, false, bv1.v, (short)0, acc11, false, false);

    __syncthreads();                       // all reads of `cur` done before refill
  }

  // C layout: lanes 0-15 VGPR r -> M=r; lanes 16-31 -> M=8+r; N = lane&15
#pragma unroll
  for (int a = 0; a < 2; ++a) {
    const int gmBase = bm0 + mt + a * 16 + kh * 8;
#pragma unroll
    for (int r = 0; r < 8; ++r) {
      const int gm = gmBase + r;
      if (gm >= M) continue;
#pragma unroll
      for (int b = 0; b < 2; ++b) {
        const int gn = bn0 + nt + b * 16 + lm;
        if (gn >= Nc) continue;
        float v;
        if (a == 0) v = (b == 0) ? acc00[r] : acc01[r];
        else        v = (b == 0) ? acc10[r] : acc11[r];
        switch (mode) {
          case EP_F32:
            outF[(size_t)gm * ldo + gn] = v;
            break;
          case EP_BF16:
            ob0[(size_t)gm * ldo + col0 + gn] = f2bf(v);
            break;
          case EP_ADD_F32:
            outF[(size_t)gm * ldo + gn] += v + bias[gn];
            break;
          case EP_GELU_BF16:
            ob0[(size_t)gm * ldo + gn] = f2bf(gelu_exact(v + bias[gn]));
            break;
          case EP_QKV: {
            const int which = gn >> 8;        // 0=q, 1=k, 2=v
            const int h = (gn >> 5) & 7;
            const int d = gn & 31;
            if (which == 0)      ob0[((size_t)h * N_NODES + gm) * DHEAD + d] = f2bf(v);
            else if (which == 1) ob1[((size_t)h * N_NODES + gm) * DHEAD + d] = f2bf(v);
            else                 ob2[((size_t)h * DHEAD + d) * N_NODES + gm] = f2bf(v); // vT
          } break;
        }
      }
    }
  }
}

// ---------------- LayerNorm: f32 in -> bf16 out (one row / block, D=256) ----------------
__global__ __launch_bounds__(256) void ln_kernel(const float* __restrict__ x,
                                                 const float* __restrict__ g,
                                                 const float* __restrict__ b,
                                                 __bf16* __restrict__ out)
{
  __shared__ float red[256];
  const int row = blockIdx.x, t = threadIdx.x;
  const float v = x[(size_t)row * DMODEL + t];
  red[t] = v; __syncthreads();
  for (int s = 128; s > 0; s >>= 1) { if (t < s) red[t] += red[t + s]; __syncthreads(); }
  const float mu = red[0] * (1.0f / DMODEL); __syncthreads();
  const float d  = v - mu;
  red[t] = d * d; __syncthreads();
  for (int s = 128; s > 0; s >>= 1) { if (t < s) red[t] += red[t + s]; __syncthreads(); }
  const float var = red[0] * (1.0f / DMODEL);
  out[(size_t)row * DMODEL + t] = f2bf(d * rsqrtf(var + 1e-5f) * g[t] + b[t]);
}

// ---------------- masked softmax: f32 scores + adj -> bf16 probs ----------------
__global__ __launch_bounds__(256) void softmax_mask_kernel(const float* __restrict__ S,
                                                           const int* __restrict__ adj,
                                                           __bf16* __restrict__ P,
                                                           float scale)
{
  __shared__ float red[256];
  const int row = blockIdx.x, t = threadIdx.x;
  float vals[8];
  float mx = -3.0e38f;
#pragma unroll
  for (int j = 0; j < 8; ++j) {
    const int c = t + j * 256;
    float s = S[(size_t)row * N_NODES + c] * scale;
    if (adj[(size_t)row * N_NODES + c] == 0) s = -1.0e9f;
    vals[j] = s;
    mx = fmaxf(mx, s);
  }
  red[t] = mx; __syncthreads();
  for (int s = 128; s > 0; s >>= 1) { if (t < s) red[t] = fmaxf(red[t], red[t + s]); __syncthreads(); }
  mx = red[0]; __syncthreads();
  float sum = 0.f;
#pragma unroll
  for (int j = 0; j < 8; ++j) { vals[j] = __expf(vals[j] - mx); sum += vals[j]; }
  red[t] = sum; __syncthreads();
  for (int s = 128; s > 0; s >>= 1) { if (t < s) red[t] += red[t + s]; __syncthreads(); }
  const float inv = 1.0f / red[0];
#pragma unroll
  for (int j = 0; j < 8; ++j)
    P[(size_t)row * N_NODES + t + j * 256] = f2bf(vals[j] * inv);
}

// ---------------- f32 -> bf16 bulk conversion ----------------
__global__ __launch_bounds__(256) void cvt_bf16_kernel(const float* __restrict__ in,
                                                       __bf16* __restrict__ out, int n)
{
  const int i = blockIdx.x * 256 + threadIdx.x;
  if (i < n) out[i] = f2bf(in[i]);
}

// ---------------- mean pool + classifier ----------------
__global__ __launch_bounds__(256) void pool_cls_kernel(const float* __restrict__ xc,
                                                       const float* __restrict__ cw,
                                                       const float* __restrict__ cb,
                                                       float* __restrict__ out)
{
  __shared__ float g[DMODEL];
  const int t = threadIdx.x;
  float s = 0.f;
  for (int r = 0; r < N_NODES; ++r) s += xc[(size_t)r * DMODEL + t];
  g[t] = s * (1.0f / N_NODES);
  __syncthreads();
  if (t < NCLS) {
    float o = cb[t];
    for (int d = 0; d < DMODEL; ++d) o += g[d] * cw[t * DMODEL + d];
    out[t] = o;
  }
}

// ---------------- host orchestration ----------------
extern "C" void kernel_launch(void* const* d_in, const int* in_sizes, int n_in,
                              void* d_out, int out_size, void* d_ws, size_t ws_size,
                              hipStream_t stream) {
  (void)in_sizes; (void)n_in; (void)out_size; (void)ws_size;
  const float* x      = (const float*)d_in[0];
  const int*   adj    = (const int*)  d_in[1];
  const float* qkv_w  = (const float*)d_in[2];
  const float* proj_w = (const float*)d_in[3];
  const float* proj_b = (const float*)d_in[4];
  const float* ff_w1  = (const float*)d_in[5];
  const float* ff_b1  = (const float*)d_in[6];
  const float* ff_w2  = (const float*)d_in[7];
  const float* ff_b2  = (const float*)d_in[8];
  const float* ln1_g  = (const float*)d_in[9];
  const float* ln1_b  = (const float*)d_in[10];
  const float* ln2_g  = (const float*)d_in[11];
  const float* ln2_b  = (const float*)d_in[12];
  const float* cls_w  = (const float*)d_in[13];
  const float* cls_b  = (const float*)d_in[14];
  float* out = (float*)d_out;

  // workspace carve-up (256 B aligned)
  char* p = (char*)d_ws;
  size_t off = 0;
  auto carve = [&](size_t bytes) { void* r = p + off; off += (bytes + 255) & ~(size_t)255; return r; };
  const size_t nQKVW = (size_t)NLAYER * 3 * DMODEL * DMODEL;
  const size_t nPW   = (size_t)NLAYER * DMODEL * DMODEL;
  const size_t nW1   = (size_t)NLAYER * FDIM * DMODEL;
  const size_t nW2   = (size_t)NLAYER * DMODEL * FDIM;
  __bf16* wq  = (__bf16*)carve(nQKVW * 2);
  __bf16* wp  = (__bf16*)carve(nPW * 2);
  __bf16* w1  = (__bf16*)carve(nW1 * 2);
  __bf16* w2  = (__bf16*)carve(nW2 * 2);
  float*  xcur = (float*) carve((size_t)N_NODES * DMODEL * 4);
  __bf16* xn   = (__bf16*)carve((size_t)N_NODES * DMODEL * 2);
  __bf16* qb   = (__bf16*)carve((size_t)N_NODES * DMODEL * 2);   // [h][n][d]
  __bf16* kb   = (__bf16*)carve((size_t)N_NODES * DMODEL * 2);   // [h][n][d]
  __bf16* vT   = (__bf16*)carve((size_t)N_NODES * DMODEL * 2);   // [h][d][n]
  __bf16* ao   = (__bf16*)carve((size_t)N_NODES * DMODEL * 2);   // attn out [n][h*32+d]
  __bf16* hm   = (__bf16*)carve((size_t)N_NODES * FDIM * 2);
  float*  sc   = (float*) carve((size_t)N_NODES * N_NODES * 4);
  __bf16* pr   = (__bf16*)carve((size_t)N_NODES * N_NODES * 2);

  const dim3 blk(256);

  // weights -> bf16 (once per call; deterministic)
  cvt_bf16_kernel<<<dim3((nQKVW + 255) / 256), blk, 0, stream>>>(qkv_w,  wq, (int)nQKVW);
  cvt_bf16_kernel<<<dim3((nPW   + 255) / 256), blk, 0, stream>>>(proj_w, wp, (int)nPW);
  cvt_bf16_kernel<<<dim3((nW1   + 255) / 256), blk, 0, stream>>>(ff_w1,  w1, (int)nW1);
  cvt_bf16_kernel<<<dim3((nW2   + 255) / 256), blk, 0, stream>>>(ff_w2,  w2, (int)nW2);

  (void)hipMemcpyAsync(xcur, x, (size_t)N_NODES * DMODEL * 4, hipMemcpyDeviceToDevice, stream);

  const float scale = 1.0f / sqrtf((float)DHEAD);
  const int mtiles = N_NODES / TM;   // 16

  for (int i = 0; i < NLAYER; ++i) {
    // LN1 -> xn (bf16)
    ln_kernel<<<dim3(N_NODES), blk, 0, stream>>>(xcur, ln1_g + i * DMODEL, ln1_b + i * DMODEL, xn);

    // QKV: (2048x256) @ (768x256)^T, scatter into q/k/vT
    gemm_nt_bf16_wmma<<<dim3((3 * DMODEL) / TN, mtiles), blk, 0, stream>>>(
        xn, wq + (size_t)i * 3 * DMODEL * DMODEL,
        N_NODES, 3 * DMODEL, DMODEL, EP_QKV,
        nullptr, nullptr, 0, 0, qb, kb, vT);

    for (int h = 0; h < NHEAD; ++h) {
      // scores = q[h] @ k[h]^T  (K = 32 -> single WMMA K-step)
      gemm_nt_bf16_wmma<<<dim3(N_NODES / TN, mtiles), blk, 0, stream>>>(
          qb + (size_t)h * N_NODES * DHEAD, kb + (size_t)h * N_NODES * DHEAD,
          N_NODES, N_NODES, DHEAD, EP_F32,
          sc, nullptr, N_NODES, 0, nullptr, nullptr, nullptr);
      // masked softmax -> probs (bf16)
      softmax_mask_kernel<<<dim3(N_NODES), blk, 0, stream>>>(sc, adj, pr, scale);
      // out[h] = probs @ v[h] = probs @ vT[h]^T  (Nc = 32)
      gemm_nt_bf16_wmma<<<dim3(1, mtiles), blk, 0, stream>>>(
          pr, vT + (size_t)h * DHEAD * N_NODES,
          N_NODES, DHEAD, N_NODES, EP_BF16,
          nullptr, nullptr, DMODEL, h * DHEAD, ao, nullptr, nullptr);
    }

    // x += ao @ proj_w^T + proj_b
    gemm_nt_bf16_wmma<<<dim3(DMODEL / TN, mtiles), blk, 0, stream>>>(
        ao, wp + (size_t)i * DMODEL * DMODEL,
        N_NODES, DMODEL, DMODEL, EP_ADD_F32,
        xcur, proj_b + i * DMODEL, DMODEL, 0, nullptr, nullptr, nullptr);

    // LN2 -> xn
    ln_kernel<<<dim3(N_NODES), blk, 0, stream>>>(xcur, ln2_g + i * DMODEL, ln2_b + i * DMODEL, xn);

    // hmid = gelu(xn @ ff_w1^T + b1) (bf16)
    gemm_nt_bf16_wmma<<<dim3(FDIM / TN, mtiles), blk, 0, stream>>>(
        xn, w1 + (size_t)i * FDIM * DMODEL,
        N_NODES, FDIM, DMODEL, EP_GELU_BF16,
        nullptr, ff_b1 + i * FDIM, FDIM, 0, hm, nullptr, nullptr);

    // x += hmid @ ff_w2^T + b2
    gemm_nt_bf16_wmma<<<dim3(DMODEL / TN, mtiles), blk, 0, stream>>>(
        hm, w2 + (size_t)i * DMODEL * FDIM,
        N_NODES, DMODEL, FDIM, EP_ADD_F32,
        xcur, ff_b2 + i * DMODEL, DMODEL, 0, nullptr, nullptr, nullptr);
  }

  pool_cls_kernel<<<dim3(1), blk, 0, stream>>>(xcur, cls_w, cls_b, out);
}
